// UniEmbedding_69715909148777
// MI455X (gfx1250) — compile-verified, matrix operand order checked by reading, not run on hardware
//
#include <hip/hip_runtime.h>

#define BB 32
#define SS 200
#define HH 64
#define NHEAD 4
#define HDIM 16
#define DFF 256
#define DTXT 768
#define NEXP 8
#define NTOK (BB * SS)

typedef __attribute__((ext_vector_type(16))) __bf16 v16bf;
typedef __attribute__((ext_vector_type(8)))  __bf16 v8bf;
typedef __attribute__((ext_vector_type(8)))  float  v8f;

union BF16x16 { v16bf v; v8bf h[2]; };

__device__ __forceinline__ __bf16 f32_to_bf16(float f) {
  unsigned u = __builtin_bit_cast(unsigned, f);
  unsigned r = u + 0x7FFFu + ((u >> 16) & 1u);
  unsigned short hs = (unsigned short)(r >> 16);
  return __builtin_bit_cast(__bf16, hs);
}

// ---------------------------------------------------------------------------
// Generic f32 -> bf16 conversion (weights)
// ---------------------------------------------------------------------------
__global__ void conv_bf16_kernel(const float* __restrict__ src,
                                 __bf16* __restrict__ dst, int n) {
  for (int i = blockIdx.x * blockDim.x + threadIdx.x; i < n;
       i += gridDim.x * blockDim.x)
    dst[i] = f32_to_bf16(src[i]);
}

// ---------------------------------------------------------------------------
// Embedding gathers
// ---------------------------------------------------------------------------
__global__ __launch_bounds__(64)
void embed_x_kernel(const int* __restrict__ hist,
                    const float* __restrict__ table,  // [NITEMS,64]
                    float* __restrict__ x) {          // [NTOK,64]
  int n = blockIdx.x;
  int id = hist[n];
  x[(size_t)n * HH + threadIdx.x] = table[(size_t)id * HH + threadIdx.x];
}

__global__ __launch_bounds__(256)
void gather_txt_kernel(const int* __restrict__ hist,
                       const float* __restrict__ table,  // [NITEMS,768]
                       __bf16* __restrict__ txt) {       // [NTOK,768]
  int n = blockIdx.x;
  int id = hist[n];
  const float* row = table + (size_t)id * DTXT;
  __bf16* orow = txt + (size_t)n * DTXT;
  for (int d = threadIdx.x; d < DTXT; d += 256)
    orow[d] = f32_to_bf16(row[d]);
}

// ---------------------------------------------------------------------------
// Transformer (f32 scalar path; ~3% of FLOPs, LN-precision sensitive)
// ---------------------------------------------------------------------------
__global__ __launch_bounds__(192)
void qkv_kernel(const float* __restrict__ x, const float* __restrict__ W,  // [192,64]
                const float* __restrict__ b, float* __restrict__ qkv) {
  __shared__ float xs[HH];
  int n = blockIdx.x;
  if (threadIdx.x < HH) xs[threadIdx.x] = x[(size_t)n * HH + threadIdx.x];
  __syncthreads();
  int o = threadIdx.x;
  const float* wr = W + (size_t)o * HH;
  float acc = b[o];
#pragma unroll 8
  for (int k = 0; k < HH; ++k) acc += xs[k] * wr[k];
  qkv[(size_t)n * (3 * HH) + o] = acc;
}

__global__ __launch_bounds__(256)
void attn_kernel(const float* __restrict__ qkv, const int* __restrict__ hist,
                 float* __restrict__ o) {
  __shared__ float Ks[SS * HDIM];
  __shared__ float Vs[SS * HDIM];
  __shared__ float kbias[SS];
  int b = blockIdx.x / NHEAD, h = blockIdx.x % NHEAD;
  for (int idx = threadIdx.x; idx < SS * HDIM; idx += 256) {
    int s = idx >> 4, d = idx & 15;
    const float* row = qkv + (size_t)(b * SS + s) * (3 * HH);
    Ks[idx] = row[HH + h * HDIM + d];
    Vs[idx] = row[2 * HH + h * HDIM + d];
  }
  for (int s = threadIdx.x; s < SS; s += 256)
    kbias[s] = (hist[b * SS + s] == 0) ? -1e9f : 0.f;
  __syncthreads();
  const float inv = 0.25f;  // 1/sqrt(16)
  for (int q = threadIdx.x; q < SS; q += 256) {
    const float* qr = qkv + (size_t)(b * SS + q) * (3 * HH) + h * HDIM;
    float qv[HDIM];
#pragma unroll
    for (int d = 0; d < HDIM; ++d) qv[d] = qr[d];
    float mx = -1e30f;
    for (int s = 0; s < SS; ++s) {
      float t = 0.f;
#pragma unroll
      for (int d = 0; d < HDIM; ++d) t += qv[d] * Ks[s * HDIM + d];
      t = t * inv + kbias[s];
      if (t > mx) mx = t;
    }
    float sum = 0.f, acc[HDIM];
#pragma unroll
    for (int d = 0; d < HDIM; ++d) acc[d] = 0.f;
    for (int s = 0; s < SS; ++s) {
      float t = 0.f;
#pragma unroll
      for (int d = 0; d < HDIM; ++d) t += qv[d] * Ks[s * HDIM + d];
      t = expf(t * inv + kbias[s] - mx);
      sum += t;
#pragma unroll
      for (int d = 0; d < HDIM; ++d) acc[d] += t * Vs[s * HDIM + d];
    }
    float* orow = o + (size_t)(b * SS + q) * HH + h * HDIM;
    float rs = 1.f / sum;
#pragma unroll
    for (int d = 0; d < HDIM; ++d) orow[d] = acc[d] * rs;
  }
}

// y = LN(x + (a @ W^T + bias)); W is [64,K]; writes back into x
__global__ __launch_bounds__(64)
void proj_res_ln_kernel(const float* __restrict__ a, const float* __restrict__ W,
                        const float* __restrict__ bias, const float* __restrict__ g,
                        const float* __restrict__ beta, float* __restrict__ x, int K) {
  __shared__ float as[DFF];
  __shared__ float sbuf[HH];
  __shared__ float smean, svar;
  int n = blockIdx.x;
  for (int k = threadIdx.x; k < K; k += 64) as[k] = a[(size_t)n * K + k];
  __syncthreads();
  int o = threadIdx.x;
  const float* wr = W + (size_t)o * K;
  float acc = bias[o];
  for (int k = 0; k < K; ++k) acc += as[k] * wr[k];
  float v = x[(size_t)n * HH + o] + acc;
  sbuf[o] = v;
  __syncthreads();
  if (o == 0) {
    float s = 0.f;
    for (int i = 0; i < HH; ++i) s += sbuf[i];
    float m = s / HH;
    float q = 0.f;
    for (int i = 0; i < HH; ++i) { float d = sbuf[i] - m; q += d * d; }
    smean = m; svar = q / HH;
  }
  __syncthreads();
  x[(size_t)n * HH + o] = (v - smean) * rsqrtf(svar + 1e-5f) * g[o] + beta[o];
}

__global__ __launch_bounds__(256)
void ff1_kernel(const float* __restrict__ x, const float* __restrict__ W,  // [256,64]
                const float* __restrict__ b, float* __restrict__ out) {
  __shared__ float xs[HH];
  int n = blockIdx.x;
  if (threadIdx.x < HH) xs[threadIdx.x] = x[(size_t)n * HH + threadIdx.x];
  __syncthreads();
  int o = threadIdx.x;
  const float* wr = W + (size_t)o * HH;
  float acc = b[o];
#pragma unroll 8
  for (int k = 0; k < HH; ++k) acc += xs[k] * wr[k];
  out[(size_t)n * DFF + o] = fmaxf(acc, 0.f);
}

// ---------------------------------------------------------------------------
// User pooling + L2 normalize  (faithful to source: keeps PADDED positions)
// ---------------------------------------------------------------------------
__global__ __launch_bounds__(64)
void user_kernel(const float* __restrict__ x, const int* __restrict__ hist,
                 const float* __restrict__ slen, float* __restrict__ user) {
  __shared__ float sbuf[HH];
  __shared__ float nrm;
  int b = blockIdx.x, h = threadIdx.x;
  float s = 0.f;
  for (int t = 0; t < SS; ++t)
    if (hist[b * SS + t] == 0) s += x[(size_t)(b * SS + t) * HH + h];
  s /= slen[b];
  sbuf[h] = s * s;
  __syncthreads();
  if (h == 0) {
    float t = 0.f;
    for (int i = 0; i < HH; ++i) t += sbuf[i];
    nrm = fmaxf(sqrtf(t), 1e-12f);
  }
  __syncthreads();
  user[b * HH + h] = s / nrm;
}

// P[b,i,j] = mean_m( dot(user[b], up_w[(i*64+j)*4+m]) + up_b[...] )
__global__ __launch_bounds__(256)
void pmat_kernel(const float* __restrict__ user, const float* __restrict__ upw,
                 const float* __restrict__ upb, float* __restrict__ P) {
  int idx = blockIdx.x * 256 + threadIdx.x;
  if (idx >= BB * HH * HH) return;
  int b = idx >> 12, k = idx & 4095;
  const float* u = user + b * HH;
  float acc = 0.f;
#pragma unroll
  for (int m = 0; m < 4; ++m) {
    const float* wr = upw + (size_t)(k * 4 + m) * HH;
    float d = upb[k * 4 + m];
    for (int t = 0; t < HH; ++t) d += u[t] * wr[t];
    acc += d;
  }
  P[idx] = acc * 0.25f;
}

// ---------------------------------------------------------------------------
// WMMA bf16 GEMM, 64x32 output tile per wave (4x2 register blocking):
//   out = relu((A[M,K] @ W[N,K]^T + bias) * scale)
// 8 WMMAs per 12 b128 loads; halves the L2 weight re-stream vs 32-row tiles.
// MODE 0: store bf16        MODE 1: store f32
// MODE 2: tm[m,n] (+)= gate[m,e] * result   (fused expert combine, N==64)
// ---------------------------------------------------------------------------
#define WMMA_BF16(A_, B_, C_)                                              \
  __builtin_amdgcn_wmma_f32_16x16x32_bf16(false, (A_), false, (B_),        \
                                          (short)0, (C_), false, false)

template <int MODE>
__global__ __launch_bounds__(256)
void wmma_gemm_kernel(const __bf16* __restrict__ A, const __bf16* __restrict__ W,
                      const float* __restrict__ bias, void* __restrict__ out,
                      int M, int K, int N, float scale,
                      const float* __restrict__ gate, int eidx, int first) {
  const int wave = threadIdx.x >> 5;
  const int lane = threadIdx.x & 31;
  const int ntn = N >> 5;  // 32-col tiles
  const int mtn = M >> 6;  // 64-row tiles
  int tile = blockIdx.x * 8 + wave;
  if (tile >= mtn * ntn) return;  // wave-uniform exit: EXEC stays all-1
  int mt = tile / ntn, nt = tile % ntn;
  const int r = lane & 15;
  const int kh = (lane >> 4) << 3;  // K half-chunk select per ISA bf16 layout
  const __bf16* ap = A + (size_t)(mt * 64 + r) * K + kh;
  const __bf16* bp = W + (size_t)(nt * 32 + r) * K + kh;
  const size_t K16 = (size_t)16 * K;
  v8f c[4][2] = {};
  for (int k0 = 0; k0 < K; k0 += 32) {
    BF16x16 af[4], bf[2];
#pragma unroll
    for (int i = 0; i < 4; ++i) {
      af[i].h[0] = *(const v8bf*)(ap + (size_t)i * K16 + k0);
      af[i].h[1] = *(const v8bf*)(ap + (size_t)i * K16 + k0 + 16);
    }
#pragma unroll
    for (int j = 0; j < 2; ++j) {
      bf[j].h[0] = *(const v8bf*)(bp + (size_t)j * K16 + k0);
      bf[j].h[1] = *(const v8bf*)(bp + (size_t)j * K16 + k0 + 16);
    }
#pragma unroll
    for (int i = 0; i < 4; ++i)
#pragma unroll
      for (int j = 0; j < 2; ++j)
        c[i][j] = WMMA_BF16(af[i].v, bf[j].v, c[i][j]);
  }
  // Epilogue. C/D layout: lane 0-15 col=lane rows 0-7; lane 16-31 col=lane-16 rows 8-15
  int n0 = nt * 32 + r;
  int mrow = mt * 64 + ((lane >> 4) << 3);
#pragma unroll
  for (int j = 0; j < 2; ++j) {
    int n = n0 + 16 * j;
    float bn = bias[n];
#pragma unroll
    for (int i = 0; i < 4; ++i) {
      int mb = mrow + 16 * i;
#pragma unroll
      for (int t = 0; t < 8; ++t) {
        float v = (c[i][j][t] + bn) * scale;
        v = v > 0.f ? v : 0.f;
        size_t oi = (size_t)(mb + t) * N + n;
        if (MODE == 0) {
          ((__bf16*)out)[oi] = f32_to_bf16(v);
        } else if (MODE == 1) {
          ((float*)out)[oi] = v;
        } else {  // fused MoE combine: tm[m,n] (+)= gate[m,e] * v
          float gv = gate[(size_t)(mb + t) * NEXP + eidx];
          float* o = (float*)out;
          if (first) o[oi] = gv * v;
          else       o[oi] += gv * v;
        }
      }
    }
  }
}

// ---------------------------------------------------------------------------
// Gate tail: 64 -> 8, relu*BN, softmax over 8 experts
// ---------------------------------------------------------------------------
__global__ __launch_bounds__(256)
void gate_final_kernel(const __bf16* __restrict__ g1, const float* __restrict__ w2,
                       const float* __restrict__ b2, float* __restrict__ gout,
                       float scale) {
  int n = blockIdx.x * 256 + threadIdx.x;
  if (n >= NTOK) return;
  float a[HH];
#pragma unroll 8
  for (int t = 0; t < HH; ++t) a[t] = (float)g1[(size_t)n * HH + t];
  float e[NEXP], mx = -1e30f;
#pragma unroll
  for (int o = 0; o < NEXP; ++o) {
    float acc = b2[o];
    for (int t = 0; t < HH; ++t) acc += a[t] * w2[o * HH + t];
    acc *= scale;
    acc = acc > 0.f ? acc : 0.f;
    e[o] = acc;
    if (acc > mx) mx = acc;
  }
  float sum = 0.f;
#pragma unroll
  for (int o = 0; o < NEXP; ++o) { e[o] = expf(e[o] - mx); sum += e[o]; }
  float rs = 1.f / sum;
#pragma unroll
  for (int o = 0; o < NEXP; ++o) gout[(size_t)n * NEXP + o] = e[o] * rs;
}

// uv[b,s,i] = sum_j P[b,i,j] * tm[b,s,j]
__global__ __launch_bounds__(64)
void uv_kernel(const float* __restrict__ P, const float* __restrict__ tm,
               float* __restrict__ uv) {
  __shared__ float ts[HH];
  int n = blockIdx.x;
  int b = n / SS;
  ts[threadIdx.x] = tm[(size_t)n * HH + threadIdx.x];
  __syncthreads();
  int i = threadIdx.x;
  const float* pr = P + ((size_t)b * HH + i) * HH;
  float acc = 0.f;
#pragma unroll 8
  for (int j = 0; j < HH; ++j) acc += pr[j] * ts[j];
  uv[(size_t)n * HH + i] = acc;
}

// ---------------------------------------------------------------------------
extern "C" void kernel_launch(void* const* d_in, const int* in_sizes, int n_in,
                              void* d_out, int out_size, void* d_ws, size_t ws_size,
                              hipStream_t stream) {
  (void)in_sizes; (void)n_in; (void)out_size; (void)ws_size;
  const float BN_S = 0.9999950000374997f;  // 1/sqrt(1+1e-5)

  const int*   hist       = (const int*)  d_in[0];
  const float* seq_len    = (const float*)d_in[1];
  const float* item_table = (const float*)d_in[2];
  const float* text_table = (const float*)d_in[3];
  const float* attn_in_w  = (const float*)d_in[4];
  const float* attn_in_b  = (const float*)d_in[5];
  const float* attn_out_w = (const float*)d_in[6];
  const float* attn_out_b = (const float*)d_in[7];
  const float* ln1_g      = (const float*)d_in[8];
  const float* ln1_b      = (const float*)d_in[9];
  const float* ff1_w      = (const float*)d_in[10];
  const float* ff1_b      = (const float*)d_in[11];
  const float* ff2_w      = (const float*)d_in[12];
  const float* ff2_b      = (const float*)d_in[13];
  const float* ln2_g      = (const float*)d_in[14];
  const float* ln2_b      = (const float*)d_in[15];
  const float* up_w       = (const float*)d_in[16];
  const float* up_b       = (const float*)d_in[17];
  const float* gate_w0    = (const float*)d_in[18];
  const float* gate_b0    = (const float*)d_in[19];
  const float* gate_w1    = (const float*)d_in[20];
  const float* gate_b1    = (const float*)d_in[21];
  const float* gate_w2    = (const float*)d_in[22];
  const float* gate_b2    = (const float*)d_in[23];
  const float* exp_w0     = (const float*)d_in[24];
  const float* exp_b0     = (const float*)d_in[25];
  const float* exp_w1     = (const float*)d_in[26];
  const float* exp_b1     = (const float*)d_in[27];
  const float* exp_w2     = (const float*)d_in[28];
  const float* exp_b2     = (const float*)d_in[29];
  const float* exp_w3     = (const float*)d_in[30];
  const float* exp_b3     = (const float*)d_in[31];

  float* out_tm = (float*)d_out;  // text_mediate [32,200,64]
  float* out_uv = out_tm + (size_t)NTOK * HH;

  // bump allocator over workspace
  char* wsb = (char*)d_ws;
  size_t off = 0;
  auto alloc = [&](size_t bytes) -> char* {
    char* p = wsb + off;
    off += (bytes + 255) & ~(size_t)255;
    return p;
  };
  float*  x     = (float*) alloc((size_t)NTOK * HH * 4);
  float*  qkv   = (float*) alloc((size_t)NTOK * 3 * HH * 4);
  float*  ao    = (float*) alloc((size_t)NTOK * HH * 4);
  float*  ff1a  = (float*) alloc((size_t)NTOK * DFF * 4);
  float*  user  = (float*) alloc((size_t)BB * HH * 4);
  float*  P     = (float*) alloc((size_t)BB * HH * HH * 4);
  __bf16* txtbf = (__bf16*)alloc((size_t)NTOK * DTXT * 2);
  __bf16* w0bf  = (__bf16*)alloc((size_t)NEXP * 512 * DTXT * 2);
  __bf16* w1bf  = (__bf16*)alloc((size_t)NEXP * 256 * 512 * 2);
  __bf16* w2bf  = (__bf16*)alloc((size_t)NEXP * 128 * 256 * 2);
  __bf16* w3bf  = (__bf16*)alloc((size_t)NEXP * 64 * 128 * 2);
  __bf16* g0wbf = (__bf16*)alloc((size_t)128 * DTXT * 2);
  __bf16* g1wbf = (__bf16*)alloc((size_t)64 * 128 * 2);
  __bf16* g0a   = (__bf16*)alloc((size_t)NTOK * 128 * 2);
  __bf16* g1a   = (__bf16*)alloc((size_t)NTOK * 64 * 2);
  float*  gsm   = (float*) alloc((size_t)NTOK * NEXP * 4);
  __bf16* a0    = (__bf16*)alloc((size_t)NTOK * 512 * 2);
  __bf16* a1    = (__bf16*)alloc((size_t)NTOK * 256 * 2);
  __bf16* a2    = (__bf16*)alloc((size_t)NTOK * 128 * 2);

  auto gblocks = [](int M, int N) { return ((M >> 6) * (N >> 5) + 7) / 8; };

  // --- embeddings ---
  embed_x_kernel<<<NTOK, 64, 0, stream>>>(hist, item_table, x);
  gather_txt_kernel<<<NTOK, 256, 0, stream>>>(hist, text_table, txtbf);

  // --- weight conversions to bf16 ---
  conv_bf16_kernel<<<4096, 256, 0, stream>>>(exp_w0, w0bf, NEXP * 512 * DTXT);
  conv_bf16_kernel<<<4096, 256, 0, stream>>>(exp_w1, w1bf, NEXP * 256 * 512);
  conv_bf16_kernel<<<1024, 256, 0, stream>>>(exp_w2, w2bf, NEXP * 128 * 256);
  conv_bf16_kernel<<<256,  256, 0, stream>>>(exp_w3, w3bf, NEXP * 64 * 128);
  conv_bf16_kernel<<<384,  256, 0, stream>>>(gate_w0, g0wbf, 128 * DTXT);
  conv_bf16_kernel<<<32,   256, 0, stream>>>(gate_w1, g1wbf, 64 * 128);

  // --- transformer (2 post-norm layers) ---
  for (int i = 0; i < 2; ++i) {
    qkv_kernel<<<NTOK, 192, 0, stream>>>(x, attn_in_w + (size_t)i * 3 * HH * HH,
                                         attn_in_b + i * 3 * HH, qkv);
    attn_kernel<<<BB * NHEAD, 256, 0, stream>>>(qkv, hist, ao);
    proj_res_ln_kernel<<<NTOK, 64, 0, stream>>>(
        ao, attn_out_w + (size_t)i * HH * HH, attn_out_b + i * HH,
        ln1_g + i * HH, ln1_b + i * HH, x, HH);
    ff1_kernel<<<NTOK, 256, 0, stream>>>(x, ff1_w + (size_t)i * DFF * HH,
                                         ff1_b + i * DFF, ff1a);
    proj_res_ln_kernel<<<NTOK, 64, 0, stream>>>(
        ff1a, ff2_w + (size_t)i * HH * DFF, ff2_b + i * HH,
        ln2_g + i * HH, ln2_b + i * HH, x, DFF);
  }

  // --- user embedding + P matrix ---
  user_kernel<<<BB, 64, 0, stream>>>(x, hist, seq_len, user);
  pmat_kernel<<<(BB * HH * HH + 255) / 256, 256, 0, stream>>>(user, up_w, up_b, P);

  // --- gate MLP (WMMA) ---
  wmma_gemm_kernel<0><<<gblocks(NTOK, 128), 256, 0, stream>>>(
      txtbf, g0wbf, gate_b0, g0a, NTOK, DTXT, 128, BN_S, nullptr, 0, 0);
  wmma_gemm_kernel<0><<<gblocks(NTOK, 64), 256, 0, stream>>>(
      g0a, g1wbf, gate_b1, g1a, NTOK, 128, 64, BN_S, nullptr, 0, 0);
  gate_final_kernel<<<(NTOK + 255) / 256, 256, 0, stream>>>(g1a, gate_w2, gate_b2,
                                                            gsm, BN_S);

  // --- experts (WMMA chain); final layer fused with gate-weighted combine ---
  for (int e = 0; e < NEXP; ++e) {
    wmma_gemm_kernel<0><<<gblocks(NTOK, 512), 256, 0, stream>>>(
        txtbf, w0bf + (size_t)e * 512 * DTXT, exp_b0 + e * 512, a0,
        NTOK, DTXT, 512, BN_S, nullptr, 0, 0);
    wmma_gemm_kernel<0><<<gblocks(NTOK, 256), 256, 0, stream>>>(
        a0, w1bf + (size_t)e * 256 * 512, exp_b1 + e * 256, a1,
        NTOK, 512, 256, BN_S, nullptr, 0, 0);
    wmma_gemm_kernel<0><<<gblocks(NTOK, 128), 256, 0, stream>>>(
        a1, w2bf + (size_t)e * 128 * 256, exp_b2 + e * 128, a2,
        NTOK, 256, 128, BN_S, nullptr, 0, 0);
    wmma_gemm_kernel<2><<<gblocks(NTOK, 64), 256, 0, stream>>>(
        a2, w3bf + (size_t)e * 64 * 128, exp_b3 + e * 64, out_tm,
        NTOK, 128, 64, BN_S, gsm, e, e == 0 ? 1 : 0);
  }

  // --- user_viewed_emb = P @ text_mediate ---
  uv_kernel<<<NTOK, 64, 0, stream>>>(P, out_tm, out_uv);
}
